// TaylorLayer_29059748725460
// MI455X (gfx1250) — matrix operand order repaired and checked by reading, not compiled, not run on hardware
//
#include <hip/hip_runtime.h>
#include <hip/hip_bf16.h>
#include <math.h>

// ---- problem constants (from reference) ----
#define NB     16384   // batch
#define NF     16      // features
#define MTOT   4844    // total monomials (deg 1..4)
#define MLOW   968     // terms of degree <= 3 (LDS resident)
#define OUTD   256     // output dim
#define NKC    31      // K chunks of 32 covering MLOW: 31*32 = 992
#define EPDS   969     // LDS row stride for epd (969 % 64 = 9 -> conflict free)

typedef __attribute__((ext_vector_type(16))) _Float16 v16h;
typedef __attribute__((ext_vector_type(8)))  float    v8f;

// ------------------------------------------------------------------
// Compile-time schedule: pass-3 bookkeeping of the reference.
//   id3[j]    : start of j's degree-3 source range (pass-3 reads p in [id3[j],968))
//   id4[j]    : monomial index where j's degree-4 block starts
//   cstart[j] : first 32-wide K-chunk covering [id3[j],968)
// Segment list: 31 unscaled segments (deg<=3 weights) followed by, for
// each j, chunks cstart[j]..30 of the x_j-scaled weight slice.
// ------------------------------------------------------------------
struct Sched { int id3[16]; int id4[16]; int cstart[16]; int nseg; };

constexpr Sched make_sched() {
    Sched s{};
    int Id[16] = {};
    for (int i = 0; i < 16; ++i) Id[i] = i;
    int Mc = 16;
    for (int pass = 0; pass < 2; ++pass) {        // deg-2 and deg-3 passes
        int hi = Id[15];
        for (int j = 0; j < 16; ++j) {
            int lo = Id[j];
            Id[j] = Mc;
            Mc += hi - lo + 1;
        }
    }
    int nseg = NKC;
    for (int j = 0; j < 16; ++j) {
        s.id3[j]    = Id[j];
        s.cstart[j] = Id[j] >> 5;
        nseg += NKC - s.cstart[j];
    }
    for (int j = 0; j < 16; ++j)                   // deg-4 block starts
        s.id4[j] = (j == 0) ? MLOW : s.id4[j - 1] + (MLOW - s.id3[j - 1]);
    s.nseg = nseg;
    return s;
}
constexpr Sched SCH = make_sched();
constexpr int   NSEG = SCH.nseg;                   // = 171

struct SegTab { short j[NSEG]; short kc[NSEG]; };
constexpr SegTab make_segtab() {
    SegTab t{};
    int s = 0;
    for (int kc = 0; kc < NKC; ++kc) { t.j[s] = -1; t.kc[s] = (short)kc; ++s; }
    for (int j = 0; j < 16; ++j)
        for (int kc = SCH.cstart[j]; kc < NKC; ++kc) { t.j[s] = (short)j; t.kc[s] = (short)kc; ++s; }
    return t;
}
constexpr SegTab SEGT = make_segtab();

// ---- LDS phase-overlapped layout (bytes) ----
#define EPD_BYTES   (16 * EPDS * 4)                 // 62,016
#define ASWZ_OFF    EPD_BYTES                       // aliases jp table
#define ASWZ_BYTES  (NKC * 32 * 16 * 2)             // 31,744
#define XLS_OFF     (ASWZ_OFF + ASWZ_BYTES)         // 93,760
#define XLS_BYTES   (16 * 17 * 4)                   // 1,088
#define SMEM_BYTES  (XLS_OFF + XLS_BYTES)           // 94,848

// ------------------------------------------------------------------
// Kernel 1: bake W (f32 [256][4844]) into segment-ordered WMMA-B f16.
// Wb[ot(16)][seg(NSEG)][lane(32)][h(16)]; 16-bit B 32x16, this code's contiguous k order (same as A):
// lane col N = lane&15, K = (lane>>4)*16 + h  ->  p = kc*32 + K.
//   seg j<0 : deg<=3 weights, W[o][p]        (0 if p>=968)
//   seg j>=0: W[o][id4[j] + p - id3[j]]      (0 outside [id3[j],968))
// ------------------------------------------------------------------
__global__ __launch_bounds__(256) void build_wb_kernel(
    const float* __restrict__ W, _Float16* __restrict__ Wb)
{
    int idx  = blockIdx.x * 256 + threadIdx.x;     // < 16*NSEG*512
    int h    = idx & 15;
    int lane = (idx >> 4) & 31;
    int seg  = (idx >> 9) % NSEG;
    int ot   = idx / (512 * NSEG);
    int o    = ot * 16 + (lane & 15);
    int j    = SEGT.j[seg];
    int p    = SEGT.kc[seg] * 32 + ((lane >> 4) << 4) + h;

    float v = 0.0f;
    if (j < 0) {
        if (p < MLOW) v = W[o * MTOT + p];
    } else {
        if (p >= SCH.id3[j] && p < MLOW)
            v = W[o * MTOT + (SCH.id4[j] + p - SCH.id3[j])];
    }
    Wb[idx] = (_Float16)v;
}

// ------------------------------------------------------------------
// Kernel 2: fused tanh -> expansion (deg<=3 in LDS) -> rank-structured
// WMMA GEMM (deg-4 = per-row x_j scaling of shared A fragments) -> +b, ReLU.
// One workgroup per 16-row batch tile; wave w owns output tiles {w, w+8}.
// ------------------------------------------------------------------
__global__ __launch_bounds__(256) void taylor_gemm_kernel(
    const float* __restrict__ in,        // [NB][NF]
    const _Float16* __restrict__ Wb,     // baked weights
    const float* __restrict__ bias,      // [OUTD]
    float* __restrict__ out)             // [NB][OUTD]
{
    __shared__ char smem[SMEM_BYTES] __attribute__((aligned(16)));
    float*    epd  = (float*)smem;                     // 16 x EPDS f32
    unsigned* jp   = (unsigned*)(smem + ASWZ_OFF);     // setup only (968 u32)
    _Float16* aswz = (_Float16*)(smem + ASWZ_OFF);     // GEMM: A fragments
    float*    xls  = (float*)(smem + XLS_OFF);         // 16 x 17 x values

    const int tid = threadIdx.x;
    const int bt  = blockIdx.x;

    // ---- expansion index table for deg-2/3 (exact reference bookkeeping) ----
    if (tid == 0) {
        int Id[16];
        #pragma unroll
        for (int i = 0; i < 16; ++i) Id[i] = i;
        int Mc = 16;
        for (int pass = 0; pass < 2; ++pass) {
            int hi = Id[15];
            for (int j = 0; j < 16; ++j) {
                int lo = Id[j];
                Id[j] = Mc;
                for (int p = lo; p <= hi; ++p)
                    jp[Mc++] = ((unsigned)j << 16) | (unsigned)p;
            }
        }
    }
    // ---- x = tanh(in): 16 rows x 16 feats == 256 threads ----
    {
        int row = tid >> 4, f = tid & 15;
        epd[row * EPDS + f] = tanhf(in[(bt * 16 + row) * NF + f]);
    }
    __syncthreads();

    // ---- deg-2 terms: m = 16..151 ----
    for (int w = tid; w < 136 * 16; w += 256) {
        int m = 16 + (w >> 4), row = w & 15;
        unsigned e = jp[m];
        epd[row * EPDS + m] =
            epd[row * EPDS + (e >> 16)] * epd[row * EPDS + (e & 0xFFFFu)];
    }
    __syncthreads();
    // ---- deg-3 terms: m = 152..967 ----
    for (int w = tid; w < 816 * 16; w += 256) {
        int m = 152 + (w >> 4), row = w & 15;
        unsigned e = jp[m];
        epd[row * EPDS + m] =
            epd[row * EPDS + (e >> 16)] * epd[row * EPDS + (e & 0xFFFFu)];
    }
    __syncthreads();   // jp dead from here; its LDS is reused for aswz

    // ---- pre-swizzle epd3 into WMMA-A layout (f16), + x copy ----
    // A 16x32 f16 layout: lane L row=L&15, h i -> K = (L>>4)*16 + i
    for (int w = tid; w < NKC * 512; w += 256) {
        int h = w & 15, ln = (w >> 4) & 31, kc = w >> 9;
        int r = ln & 15;
        int K = kc * 32 + ((ln >> 4) << 4) + h;
        float v = (K < MLOW) ? epd[r * EPDS + K] : 0.0f;
        aswz[w] = (_Float16)v;
    }
    {
        int row = tid >> 4, f = tid & 15;
        xls[row * 17 + f] = epd[row * EPDS + f];
    }
    __syncthreads();

    // ---- WMMA GEMM over NSEG segments ----
    const int lane = tid & 31;
    const int wv   = tid >> 5;
    const int row  = lane & 15;

    const _Float16* wb0 = Wb + ((size_t)(wv       * NSEG) << 9);
    const _Float16* wb1 = Wb + ((size_t)((wv + 8) * NSEG) << 9);

    v8f c0 = {};
    v8f c1 = {};
    int seg = 0;

    // deg<=3 block: unscaled shared A fragments
    for (int kc = 0; kc < NKC; ++kc, ++seg) {
        v16h a  = *(const v16h*)(aswz + ((kc * 32 + lane) << 4));
        v16h b0 = *(const v16h*)(wb0 + (((size_t)seg * 32 + lane) << 4));
        v16h b1 = *(const v16h*)(wb1 + (((size_t)seg * 32 + lane) << 4));
        c0 = __builtin_amdgcn_wmma_f32_16x16x32_f16(false, a, false, b0,
                                                    (short)0, c0, false, false);
        c1 = __builtin_amdgcn_wmma_f32_16x16x32_f16(false, a, false, b1,
                                                    (short)0, c1, false, false);
    }
    // deg-4 blocks: same A fragments scaled by per-row x_j (lane scalar)
    for (int j = 0; j < 16; ++j) {
        const _Float16 sc = (_Float16)xls[row * 17 + j];
        for (int kc = SCH.cstart[j]; kc < NKC; ++kc, ++seg) {
            v16h a = *(const v16h*)(aswz + ((kc * 32 + lane) << 4));
            a = a * sc;                                  // v_pk_mul_f16 x8
            v16h b0 = *(const v16h*)(wb0 + (((size_t)seg * 32 + lane) << 4));
            v16h b1 = *(const v16h*)(wb1 + (((size_t)seg * 32 + lane) << 4));
            c0 = __builtin_amdgcn_wmma_f32_16x16x32_f16(false, a, false, b0,
                                                        (short)0, c0, false, false);
            c1 = __builtin_amdgcn_wmma_f32_16x16x32_f16(false, a, false, b1,
                                                        (short)0, c1, false, false);
        }
    }

    // ---- epilogue: +bias, ReLU, store (C/D layout per ISA 7.12.2) ----
    const int col   = lane & 15;
    const int rbase = (lane >> 4) * 8;
    const int o0 = wv * 16 + col;
    const int o1 = (wv + 8) * 16 + col;
    const float bia0 = bias[o0];
    const float bia1 = bias[o1];
    #pragma unroll
    for (int r = 0; r < 8; ++r) {
        int b = bt * 16 + rbase + r;
        float v0 = c0[r] + bia0;
        float v1 = c1[r] + bia1;
        out[b * OUTD + o0] = v0 > 0.0f ? v0 : 0.0f;
        out[b * OUTD + o1] = v1 > 0.0f ? v1 : 0.0f;
    }
}

// ------------------------------------------------------------------
// launch: inputs = {inputs[f32 NB*NF], W[f32 OUTD*MTOT], b[f32 OUTD]}
// workspace: baked f16 weights, 16*NSEG*512*2 = 2,801,664 bytes
// ------------------------------------------------------------------
extern "C" void kernel_launch(void* const* d_in, const int* in_sizes, int n_in,
                              void* d_out, int out_size, void* d_ws, size_t ws_size,
                              hipStream_t stream)
{
    (void)in_sizes; (void)n_in; (void)out_size; (void)ws_size;
    const float* d_x  = (const float*)d_in[0];
    const float* d_W  = (const float*)d_in[1];
    const float* d_b  = (const float*)d_in[2];
    float*       d_o  = (float*)d_out;
    _Float16*    d_wb = (_Float16*)d_ws;

    // 16*NSEG*512 elements -> NSEG*32 blocks of 256 threads
    build_wb_kernel<<<NSEG * 32, 256, 0, stream>>>(d_W, d_wb);

    // one workgroup per 16-row batch tile
    taylor_gemm_kernel<<<NB / 16, 256, 0, stream>>>(d_x, d_wb, d_b, d_o);
}
